// FrameTransformer_57389353009719
// MI455X (gfx1250) — matrix operand order repaired and checked by reading, not compiled, not run on hardware
//
#include <hip/hip_runtime.h>
#include <hip/hip_bf16.h>

// ---------------------------------------------------------------------------
// FrameTransformer block for MI455X (gfx1250, wave32, WMMA).
// All heavy math routed through v_wmma_f32_16x16x32_f16 (f16 in, f32 acc).
// 256x128x32 block tiles, 64x64 wave tiles (16 WMMA / k-step / wave),
// double-buffered LDS, fully vectorized b128 staging (weights pre-transposed).
// ---------------------------------------------------------------------------

typedef _Float16 v16h __attribute__((ext_vector_type(16)));
typedef float    v8f  __attribute__((ext_vector_type(8)));

#define BM 256
#define BN 128
#define BK 32
#define LDSP 40   // 40 halves = 80B row pitch -> 16B-aligned fragment reads

enum { GF_RELU = 1, GF_ACC = 2, GF_OUTF16 = 4, GF_TRANSC = 8 };
enum { M_KCONT = 1, M_MCONT = 2 };   // A modes: k-contiguous / m-contiguous

union Frag { v16h v; uint4 q[2]; };

// Strided/batched WMMA GEMM:
//   C[z][m][n] (+)= alpha * sum_k A[z][m*ldaM + k*ldaK] * B[z][k + n*ldbN] (+ bias[n])
// B is always k-contiguous (ldbK==1); weights are pre-transposed to [n][k].
// Batch offsets: off = (z/zdiv)*z?1 + (z%zdiv)*z?2  (element units of each buffer).
// AMODE: M_KCONT -> ldaK==1 (vector loads along k)
//        M_MCONT -> ldaM==1 (vector loads along m, LDS scatter; only er^T)
template <int AMODE>
__global__ __launch_bounds__(256)
void gemm_wmma_f16(const _Float16* __restrict__ A, long ldaM, long ldaK,
                   const _Float16* __restrict__ B, long ldbN,
                   void* __restrict__ C, long ldc,
                   int K, float alpha, const float* __restrict__ bias, int flags,
                   int zdiv, long zA1, long zA2, long zB1, long zB2,
                   long zC1, long zC2)
{
    __shared__ _Float16 As[2][BM][LDSP];   // 2*256*40*2 = 40 KB
    __shared__ _Float16 Bs[2][BN][LDSP];   // 2*128*40*2 = 20 KB

    const int z = blockIdx.z;
    const long offA = (long)(z / zdiv) * zA1 + (long)(z % zdiv) * zA2;
    const long offB = (long)(z / zdiv) * zB1 + (long)(z % zdiv) * zB2;
    const long offC = (long)(z / zdiv) * zC1 + (long)(z % zdiv) * zC2;

    const int m0 = blockIdx.y * BM;
    const int n0 = blockIdx.x * BN;
    const int tid  = threadIdx.x;
    const int lane = tid & 31;
    const int wave = tid >> 5;
    const int wm = wave & 3;   // 4 waves along M (4*64 = 256)
    const int wn = wave >> 2;  // 2 waves along N (2*64 = 128)

    // ---- per-thread staging pointers (incremental; no per-element muls) ----
    const _Float16* aptr;
    long astep;
    int ar, ac;                          // LDS coords owned by this thread
    if (AMODE == M_KCONT) {              // thread: row tid, all 32 k
        ar = tid; ac = 0;
        aptr = A + offA + (long)(m0 + ar) * ldaM;
        astep = BK;
    } else {                             // M_MCONT: thread: k tid>>3, m-seg (tid&7)*32
        ac = tid >> 3; ar = (tid & 7) * 32;
        aptr = A + offA + (long)ac * ldaK + (m0 + ar);
        astep = (long)BK * ldaK;
    }
    // B: always k-contiguous. thread: n tid>>1, k-seg (tid&1)*16
    const int br = tid >> 1, bc = (tid & 1) * 16;
    const _Float16* bptr = B + offB + (long)(n0 + br) * ldbN + bc;

    auto load_tile = [&](int buf) {
        if (AMODE == M_KCONT) {
            uint4 q0 = *(const uint4*)aptr;
            uint4 q1 = *(const uint4*)(aptr + 8);
            uint4 q2 = *(const uint4*)(aptr + 16);
            uint4 q3 = *(const uint4*)(aptr + 24);
            *(uint4*)&As[buf][ar][0]  = q0;
            *(uint4*)&As[buf][ar][8]  = q1;
            *(uint4*)&As[buf][ar][16] = q2;
            *(uint4*)&As[buf][ar][24] = q3;
        } else {
            _Float16 t[32];
            *(uint4*)(t +  0) = *(const uint4*)aptr;
            *(uint4*)(t +  8) = *(const uint4*)(aptr + 8);
            *(uint4*)(t + 16) = *(const uint4*)(aptr + 16);
            *(uint4*)(t + 24) = *(const uint4*)(aptr + 24);
#pragma unroll
            for (int j = 0; j < 32; ++j) As[buf][ar + j][ac] = t[j];
        }
        aptr += astep;
        {
            uint4 q0 = *(const uint4*)bptr;
            uint4 q1 = *(const uint4*)(bptr + 8);
            *(uint4*)&Bs[buf][br][bc]     = q0;
            *(uint4*)&Bs[buf][br][bc + 8] = q1;
        }
        bptr += BK;
    };

    v8f acc[4][4] = {};

    // ISA 7.12.2 fragment addressing (wave32):
    //  A 16x32 f16: lanes 0-15 row M=lane, K {0..7,16..23}; lanes 16-31 K {8..15,24..31}
    //  B 32x16 f16: lanes 0-15 col N=lane, K {0..7,16..23}; lanes 16-31 K {8..15,24..31}
    const int abase = (lane < 16) ? 0 : 8;
    const int arow  = wm * 64 + (lane & 15);
    const int bcol  = wn * 64 + (lane & 15);
    const int kb    = (lane >> 4) << 4;

    load_tile(0);
    int cur = 0;
    for (int k0 = 0; k0 < K; k0 += BK) {
        __syncthreads();
        if (k0 + BK < K) {
            load_tile(cur ^ 1);                 // overlaps with WMMA below
            if (k0 + 2 * BK < K) {              // gfx1250 global_prefetch_b8
                __builtin_prefetch(aptr, 0, 1);
                __builtin_prefetch(bptr, 0, 1);
            }
        }

        Frag a[4];
#pragma unroll
        for (int i = 0; i < 4; ++i) {
            const _Float16* p = &As[cur][arow + i * 16][0];
            a[i].q[0] = *(const uint4*)(p + abase);        // K base..base+7
            a[i].q[1] = *(const uint4*)(p + abase + 16);   // K base+16..base+23
        }
#pragma unroll
        for (int j = 0; j < 4; ++j) {
            Frag bf;
            const _Float16* p = &Bs[cur][bcol + j * 16][0];
            bf.q[0] = *(const uint4*)(p + abase);          // K base..base+7
            bf.q[1] = *(const uint4*)(p + abase + 16);     // K base+16..base+23
#pragma unroll
            for (int i = 0; i < 4; ++i)
                acc[i][j] = __builtin_amdgcn_wmma_f32_16x16x32_f16(
                    false, a[i].v, false, bf.v, (short)0, acc[i][j], false, false);
        }
        cur ^= 1;
    }

    // Epilogue.  C layout: vgpr v -> row v (lanes 0-15) / v+8 (lanes 16-31), col lane%16.
    const int lm  = lane & 15;
    const int lh8 = (lane >> 4) * 8;
#pragma unroll
    for (int i = 0; i < 4; ++i) {
#pragma unroll
        for (int j = 0; j < 4; ++j) {
            const int rb = m0 + wm * 64 + i * 16 + lh8;
            const int cb = n0 + wn * 64 + j * 16 + lm;
            const float badd = bias ? bias[cb] : 0.f;
#pragma unroll
            for (int v = 0; v < 8; ++v) {
                float r = acc[i][j][v] * alpha + badd;
                if (flags & GF_RELU) r = fmaxf(r, 0.f);
                const long idx = (flags & GF_TRANSC)
                                   ? offC + (long)cb * ldc + (rb + v)
                                   : offC + (long)(rb + v) * ldc + cb;
                if (flags & GF_OUTF16) {
                    ((_Float16*)C)[idx] = (_Float16)r;
                } else {
                    float* Cf = (float*)C;
                    Cf[idx] = (flags & GF_ACC) ? (Cf[idx] + r) : r;
                }
            }
        }
    }
}

// ---------------------------------------------------------------------------
// Elementwise / normalization kernels
// ---------------------------------------------------------------------------

// src [8,8,1024,512] * w_bn[8] -> xres [4096 tokens][1024], token = b*512+w
__global__ void bottleneck_kernel(const float* __restrict__ src,
                                  const float* __restrict__ wbn,
                                  float* __restrict__ xres)
{
    int idx = blockIdx.x * 256 + threadIdx.x;        // enumerates (b,h,w), w fastest
    if (idx >= 8 * 1024 * 512) return;
    int b = idx >> 19;
    int r = idx & 524287;
    int h = r >> 9;
    int w = r & 511;
    const float* s = src + (long)b * 4194304 + (long)h * 512 + w;
    float acc = 0.f;
#pragma unroll
    for (int c = 0; c < 8; ++c) acc += s[(long)c * 524288] * wbn[c];
    xres[((long)(b * 512 + w)) * 1024 + h] = acc;
}

// Row LayerNorm (fp32 in) -> f16 out.  One block (256 thr) per row.
__global__ void ln_kernel(const float* __restrict__ x,
                          const float* __restrict__ g,
                          const float* __restrict__ b,
                          _Float16* __restrict__ y, int width)
{
    __shared__ float red[256];
    const long row = blockIdx.x;
    const float* xr = x + row * width;
    const int tid = threadIdx.x;

    float s = 0.f;
    for (int i = tid; i < width; i += 256) s += xr[i];
    red[tid] = s; __syncthreads();
    for (int o = 128; o; o >>= 1) { if (tid < o) red[tid] += red[tid + o]; __syncthreads(); }
    const float mean = red[0] / width;
    __syncthreads();

    float s2 = 0.f;
    for (int i = tid; i < width; i += 256) { float d = xr[i] - mean; s2 += d * d; }
    red[tid] = s2; __syncthreads();
    for (int o = 128; o; o >>= 1) { if (tid < o) red[tid] += red[tid + o]; __syncthreads(); }
    const float rstd = rsqrtf(red[0] / width + 1e-5f);

    _Float16* yr = y + row * width;
    for (int i = tid; i < width; i += 256)
        yr[i] = (_Float16)((xr[i] - mean) * rstd * g[i] + b[i]);
}

// x += a * sigmoid(gate), hg = [T, 2048] with a|gate split
__global__ void glu_kernel(const float* __restrict__ hg, float* __restrict__ x)
{
    int idx = blockIdx.x * 256 + threadIdx.x;
    if (idx >= 4096 * 1024) return;
    int t = idx >> 10, i = idx & 1023;
    float a = hg[(long)t * 2048 + i];
    float gt = hg[(long)t * 2048 + 1024 + i];
    x[idx] += a / (1.f + __expf(-gt));
}

// im2col for k=3 pad=1 conv over [T,1024] f16 -> [T,3072] f16 (tap-major cols)
__global__ void im2col_kernel(const _Float16* __restrict__ h, _Float16* __restrict__ out)
{
    int idx = blockIdx.x * 256 + threadIdx.x;
    if (idx >= 4096 * 3072) return;
    int t = idx / 3072, r = idx % 3072;
    int tap = r >> 10, ci = r & 1023;
    int wt = t & 511;
    int w2 = wt + tap - 1;
    _Float16 v = (_Float16)0.f;
    if (w2 >= 0 && w2 < 512) v = h[(long)(t - wt + w2) * 1024 + ci];
    out[idx] = v;
}

// depthwise conv k=9 pad=4 over [T,4096] f16, weights fp32 [4096,1,9] -> f16 out
__global__ void dwconv_kernel(const _Float16* __restrict__ h,
                              const float* __restrict__ w,
                              _Float16* __restrict__ out)
{
    int idx = blockIdx.x * 256 + threadIdx.x;
    if (idx >= 4096 * 4096) return;
    int t = idx >> 12, c = idx & 4095;
    int wt = t & 511;
    float acc = 0.f;
#pragma unroll
    for (int j = 0; j < 9; ++j) {
        int ws_ = wt + j - 4;
        if (ws_ >= 0 && ws_ < 512)
            acc += (float)h[(long)(t - wt + ws_) * 4096 + c] * w[c * 9 + j];
    }
    out[idx] = (_Float16)acc;
}

// Softmax over 512-wide rows of fp32 scores; writes f16 probs IN PLACE over row start.
__global__ void softmax_kernel(float* __restrict__ scores)
{
    __shared__ float red[256];
    const long row = blockIdx.x;
    float* sr = scores + row * 512;
    _Float16* pr = (_Float16*)sr;
    const int tid = threadIdx.x;

    float v0 = sr[tid], v1 = sr[tid + 256];   // all reads before any write
    red[tid] = fmaxf(v0, v1); __syncthreads();
    for (int o = 128; o; o >>= 1) { if (tid < o) red[tid] = fmaxf(red[tid], red[tid + o]); __syncthreads(); }
    const float m = red[0]; __syncthreads();
    float e0 = __expf(v0 - m), e1 = __expf(v1 - m);
    red[tid] = e0 + e1; __syncthreads();
    for (int o = 128; o; o >>= 1) { if (tid < o) red[tid] += red[tid + o]; __syncthreads(); }
    const float inv = 1.f / red[0];
    pr[tid] = (_Float16)(e0 * inv);
    pr[tid + 256] = (_Float16)(e1 * inv);
}

__global__ void cvt_f32_f16(const float* __restrict__ s, _Float16* __restrict__ d, int n)
{
    int i = blockIdx.x * 256 + threadIdx.x;
    if (i < n) d[i] = (_Float16)s[i];
}

// row-major [K][N] fp32 -> transposed f16 [N][K]  (B operands become k-contiguous)
__global__ void pack_wT(const float* __restrict__ s, _Float16* __restrict__ d, int Kd, int Nd)
{
    int i = blockIdx.x * 256 + threadIdx.x;
    if (i >= Kd * Nd) return;
    int k = i / Nd, n = i % Nd;
    d[(long)n * Kd + k] = (_Float16)s[i];
}

// w_c1R [out=1024][in=1024][3] -> BT [out][tap*1024+ci]  (within-row permute)
__global__ void pack_c1rT(const float* __restrict__ s, _Float16* __restrict__ d)
{
    int i = blockIdx.x * 256 + threadIdx.x;
    if (i >= 1024 * 3072) return;
    int n = i / 3072, r = i % 3072;
    int ci = r / 3, tap = r % 3;
    d[(long)n * 3072 + tap * 1024 + ci] = (_Float16)s[i];
}

// xres [b*512+w][h] -> out [b][1][h][w]
__global__ void out_transpose_kernel(const float* __restrict__ x, float* __restrict__ out)
{
    int idx = blockIdx.x * 256 + threadIdx.x;
    if (idx >= 8 * 1024 * 512) return;
    int b = idx >> 19;
    int h = (idx >> 9) & 1023;
    int w = idx & 511;
    out[idx] = x[((long)(b * 512 + w)) * 1024 + h];
}

// ---------------------------------------------------------------------------
// Host launcher
// ---------------------------------------------------------------------------

extern "C" void kernel_launch(void* const* d_in, const int* in_sizes, int n_in,
                              void* d_out, int out_size, void* d_ws, size_t ws_size,
                              hipStream_t stream)
{
    (void)in_sizes; (void)n_in; (void)out_size; (void)ws_size;

    const float* src    = (const float*)d_in[0];
    const float* w_bn   = (const float*)d_in[1];
    const float* g1     = (const float*)d_in[2];
    const float* b1     = (const float*)d_in[3];
    const float* w_glu  = (const float*)d_in[4];
    const float* g2     = (const float*)d_in[5];
    const float* b2     = (const float*)d_in[6];
    const float* w_c1L  = (const float*)d_in[7];
    const float* w_c1R  = (const float*)d_in[8];
    const float* g3     = (const float*)d_in[9];
    const float* b3     = (const float*)d_in[10];
    const float* w_c2dw = (const float*)d_in[11];
    const float* w_c2pw = (const float*)d_in[12];
    const float* g4     = (const float*)d_in[13];
    const float* b4     = (const float*)d_in[14];
    const float* wq     = (const float*)d_in[15];
    const float* bq     = (const float*)d_in[16];
    const float* wk     = (const float*)d_in[17];
    const float* bk     = (const float*)d_in[18];
    const float* wv     = (const float*)d_in[19];
    const float* bv     = (const float*)d_in[20];
    const float* wo     = (const float*)d_in[21];
    const float* bo     = (const float*)d_in[22];
    const float* er     = (const float*)d_in[23];
    const float* g5     = (const float*)d_in[24];
    const float* b5     = (const float*)d_in[25];
    const float* w_c3   = (const float*)d_in[26];
    const float* w_c4   = (const float*)d_in[27];

    const int T = 4096;              // B*W tokens
    const long HB = 524288;          // 512*1024 (per-batch q/k/v row span, halves)

    // ---- workspace layout ----
    char* cur = (char*)d_ws;
    auto alloc = [&](size_t bytes) -> void* {
        void* p = (void*)cur;
        cur += (bytes + 255) & ~(size_t)255;
        return p;
    };
    float*    xres   = (float*)   alloc((size_t)T * 1024 * 4);   // residual stream
    _Float16* act    = (_Float16*)alloc((size_t)T * 4096 * 2);   // LN outputs / attn_o
    _Float16* big    = (_Float16*)alloc((size_t)T * 4096 * 2);   // im2col / dw out / qkv / mlp
    float*    tmp1   = (float*)   alloc((size_t)T * 4096 * 4);   // f32 GEMM out / scores
    _Float16* wgluT  = (_Float16*)alloc((size_t)1024 * 2048 * 2);
    _Float16* wc1LT  = (_Float16*)alloc((size_t)1024 * 4096 * 2);
    _Float16* wc1RT  = (_Float16*)alloc((size_t)3072 * 1024 * 2);
    _Float16* wc2pT  = (_Float16*)alloc((size_t)4096 * 1024 * 2);
    _Float16* wqT    = (_Float16*)alloc((size_t)1024 * 1024 * 2);
    _Float16* wkT    = (_Float16*)alloc((size_t)1024 * 1024 * 2);
    _Float16* wvT    = (_Float16*)alloc((size_t)1024 * 1024 * 2);
    _Float16* woT    = (_Float16*)alloc((size_t)1024 * 1024 * 2);
    _Float16* erH    = (_Float16*)alloc((size_t)256 * 512 * 2);
    _Float16* wc3T   = (_Float16*)alloc((size_t)1024 * 4096 * 2);
    _Float16* wc4T   = (_Float16*)alloc((size_t)4096 * 1024 * 2);

    _Float16* qb  = big;                      // q/k/vT live in `big` during attention
    _Float16* kb2 = big + (size_t)T * 1024;
    _Float16* vT  = big + (size_t)T * 1024 * 2;   // [channel][token] = [1024][4096]

    auto cdiv = [](int a, int b) { return (a + b - 1) / b; };
    auto cvt = [&](const float* s, _Float16* d, int n) {
        cvt_f32_f16<<<cdiv(n, 256), 256, 0, stream>>>(s, d, n);
    };
    auto packT = [&](const float* s, _Float16* d, int Kd, int Nd) {
        pack_wT<<<cdiv(Kd * Nd, 256), 256, 0, stream>>>(s, d, Kd, Nd);
    };
    auto gemm = [&](int amode,
                    const _Float16* A, long ldaM, long ldaK,
                    const _Float16* B, long ldbN,
                    void* C, long ldc, int M, int N, int K,
                    float alpha, const float* bias, int flags,
                    int nz = 1, int zdiv = 1,
                    long zA1 = 0, long zA2 = 0, long zB1 = 0, long zB2 = 0,
                    long zC1 = 0, long zC2 = 0) {
        dim3 grid(N / BN, M / BM, nz), block(256);
        if (amode == M_KCONT)
            gemm_wmma_f16<M_KCONT><<<grid, block, 0, stream>>>(
                A, ldaM, ldaK, B, ldbN, C, ldc, K, alpha, bias, flags,
                zdiv, zA1, zA2, zB1, zB2, zC1, zC2);
        else
            gemm_wmma_f16<M_MCONT><<<grid, block, 0, stream>>>(
                A, ldaM, ldaK, B, ldbN, C, ldc, K, alpha, bias, flags,
                zdiv, zA1, zA2, zB1, zB2, zC1, zC2);
    };

    // ---- weight prep: fp32 -> f16, transposed to [n][k] (k-contiguous B) ----
    packT(w_glu, wgluT, 1024, 2048);
    packT(w_c1L, wc1LT, 1024, 4096);
    pack_c1rT<<<cdiv(1024 * 3072, 256), 256, 0, stream>>>(w_c1R, wc1RT);
    cvt(w_c2pw, wc2pT, 4096 * 1024);          // [out][in] is already [n][k]
    packT(wq, wqT, 1024, 1024);
    packT(wk, wkT, 1024, 1024);
    packT(wv, wvT, 1024, 1024);
    packT(wo, woT, 1024, 1024);
    cvt(er, erH, 256 * 512);
    packT(w_c3, wc3T, 1024, 4096);
    packT(w_c4, wc4T, 4096, 1024);

    // ---- 0) bottleneck: src -> xres [T,1024] ----
    bottleneck_kernel<<<cdiv(8 * 1024 * 512, 256), 256, 0, stream>>>(src, w_bn, xres);

    // ---- 1) GLU block ----
    ln_kernel<<<T, 256, 0, stream>>>(xres, g1, b1, act, 1024);
    gemm(M_KCONT, act, 1024, 1, wgluT, 1024, tmp1, 2048,
         T, 2048, 1024, 1.f, nullptr, 0);
    glu_kernel<<<cdiv(T * 1024, 256), 256, 0, stream>>>(tmp1, xres);

    // ---- 2) conv feed-forward block ----
    ln_kernel<<<T, 256, 0, stream>>>(xres, g2, b2, act, 1024);
    gemm(M_KCONT, act, 1024, 1, wc1LT, 1024, tmp1, 4096,
         T, 4096, 1024, 1.f, nullptr, GF_RELU);
    im2col_kernel<<<cdiv(T * 3072, 256), 256, 0, stream>>>(act, big);
    gemm(M_KCONT, big, 3072, 1, wc1RT, 3072, tmp1, 4096,
         T, 1024, 3072, 1.f, nullptr, GF_RELU | GF_ACC);     // hL + pad(hR)
    ln_kernel<<<T, 256, 0, stream>>>(tmp1, g3, b3, act, 4096);
    dwconv_kernel<<<cdiv(T * 4096, 256), 256, 0, stream>>>(act, w_c2dw, big);
    gemm(M_KCONT, big, 4096, 1, wc2pT, 4096, xres, 1024,
         T, 1024, 4096, 1.f, nullptr, GF_ACC);

    // ---- 3) multiband attention (4 bands, d=256, 32 heads total) ----
    ln_kernel<<<T, 256, 0, stream>>>(xres, g4, b4, act, 1024);
    gemm(M_KCONT, act, 1024, 1, wqT, 1024, qb, 1024,
         T, 1024, 1024, 1.f, bq, GF_OUTF16);
    gemm(M_KCONT, act, 1024, 1, wkT, 1024, kb2, 1024,
         T, 1024, 1024, 1.f, bk, GF_OUTF16);
    gemm(M_KCONT, act, 1024, 1, wvT, 1024, vT, 4096,          // V^T [chan][tok]
         T, 1024, 1024, 1.f, bv, GF_OUTF16 | GF_TRANSC);

    const float isq = 1.f / 32.f;   // 1/sqrt(1024)
    // scores[b,n,i,j] = (1/32) * Q_i . K_j   (K^T is k-contiguous via strides)
    gemm(M_KCONT, qb, 1024, 1, kb2, 1024, tmp1, 512,
         512, 512, 256, isq, nullptr, 0,
         32, 4, HB, 256, HB, 256, 1048576, 262144);
    // scores[b,n,i,j] += (1/32) * sum_d er[d,i] * q[b,j,n*256+d]   (skewed rel-pos)
    gemm(M_MCONT, erH, 1, 512, qb, 1024, tmp1, 512,
         512, 512, 256, isq, nullptr, GF_ACC,
         32, 4, 0, 0, HB, 256, 1048576, 262144);
    softmax_kernel<<<32 * 512, 256, 0, stream>>>(tmp1);      // in-place f32 -> f16 probs
    // o[b,w,n*256+d] = sum_v probs[w,v] * V^T[n*256+d][b*512+v]
    gemm(M_KCONT, (const _Float16*)tmp1, 1024, 1, vT, 4096, act, 1024,
         512, 256, 512, 1.f, nullptr, GF_OUTF16,
         32, 4, 2097152, 524288, 512, 1048576, HB, 256);
    // x += o @ wo + bo
    gemm(M_KCONT, act, 1024, 1, woT, 1024, xres, 1024,
         T, 1024, 1024, 1.f, bo, GF_ACC);

    // ---- 4) final MLP block ----
    ln_kernel<<<T, 256, 0, stream>>>(xres, g5, b5, act, 1024);
    gemm(M_KCONT, act, 1024, 1, wc3T, 1024, big, 4096,
         T, 4096, 1024, 1.f, nullptr, GF_RELU | GF_OUTF16);
    gemm(M_KCONT, big, 4096, 1, wc4T, 4096, xres, 1024,
         T, 1024, 4096, 1.f, nullptr, GF_ACC);

    // ---- 5) output transpose [T,1024] -> [8,1,1024,512] ----
    out_transpose_kernel<<<cdiv(8 * 1024 * 512, 256), 256, 0, stream>>>(xres, (float*)d_out);
}
